// GraphGRU_75694503624994
// MI455X (gfx1250) — matrix operand order repaired and checked by reading, not compiled, not run on hardware
//
#include <hip/hip_runtime.h>
#include <hip/hip_bf16.h>
#include <math.h>

// Problem constants (match reference)
constexpr int B   = 32;
constexpr int N   = 64;
constexpr int T   = 128;
constexpr int D   = 64;
constexpr int EMB = 128;
constexpr int H   = 256;
constexpr int E   = 512;
constexpr int G3H = 3 * H;   // 768

typedef __attribute__((ext_vector_type(16))) _Float16 v16h;
typedef __attribute__((ext_vector_type(8)))  _Float16 v8h;
typedef __attribute__((ext_vector_type(8)))  float    v8f;

// ---------------------------------------------------------------------------
// WMMA helper: D = A(16x32 f16) * B(32x16 f16) + C(16x16 f32)
// ---------------------------------------------------------------------------
__device__ __forceinline__ v8f wmma_f16(v16h a, v16h b, v8f c) {
    return __builtin_amdgcn_wmma_f32_16x16x32_f16(
        /*neg_a=*/false, a, /*neg_b=*/false, b,
        /*c_mod=*/(short)0, c, /*reuse_a=*/false, /*reuse_b=*/false);
}

// A-fragment (16x32): lane l holds row m=l&15; halves 0..7 -> K=kb+i,
// halves 8..15 -> K=kb+16+(i-8), kb=(l<16)?0:8.  Two 16B-aligned f16 runs.
__device__ __forceinline__ v16h load_a_f16(const _Float16* __restrict__ base,
                                           size_t ld, int k0) {
    int l  = threadIdx.x & 31;
    int m  = l & 15;
    int kb = (l < 16) ? 0 : 8;
    const _Float16* p = base + (size_t)m * ld + (size_t)(k0 + kb);
    v8h lo = *(const v8h*)p;
    v8h hi = *(const v8h*)(p + 16);
    v16h r;
#pragma unroll
    for (int i = 0; i < 8; ++i) { r[i] = lo[i]; r[8 + i] = hi[i]; }
    return r;
}

// Same A-fragment layout but from an f32 source (used for x only).
__device__ __forceinline__ v16h load_a_f32(const float* __restrict__ base,
                                           size_t ld, int k0) {
    int l  = threadIdx.x & 31;
    int m  = l & 15;
    int kb = (l < 16) ? 0 : 8;
    const float* p = base + (size_t)m * ld + (size_t)(k0 + kb);
    v16h r;
#pragma unroll
    for (int i = 0; i < 8; ++i) r[i] = (_Float16)p[i];
#pragma unroll
    for (int i = 0; i < 8; ++i) r[8 + i] = (_Float16)p[16 + i];
    return r;
}

// A-fragment from an LDS f16 tile (row stride `ld` halves, multiple of 8).
__device__ __forceinline__ v16h load_a_lds(const _Float16* __restrict__ base,
                                           int ld, int k0) {
    int l  = threadIdx.x & 31;
    int m  = l & 15;
    int kb = (l < 16) ? 0 : 8;
    const _Float16* p = base + m * ld + k0 + kb;
    v8h lo = *(const v8h*)p;
    v8h hi = *(const v8h*)(p + 16);
    v16h r;
#pragma unroll
    for (int i = 0; i < 8; ++i) { r[i] = lo[i]; r[8 + i] = hi[i]; }
    return r;
}

// B-fragment (32x16) from f16 source laid out [col][k], K contiguous:
// lane l holds column c=l&15; halves 0..7 -> K=kb+i, 8..15 -> K=kb+16+(i-8),
// kb=(l<16)?0:8.  Two 16B runs (matches A-fragment).
__device__ __forceinline__ v16h load_b_f16(const _Float16* __restrict__ base,
                                           size_t ld, int col0, int k0) {
    int l  = threadIdx.x & 31;
    int c  = l & 15;
    int kb = (l < 16) ? 0 : 8;
    const _Float16* p = base + (size_t)(col0 + c) * ld + (size_t)(k0 + kb);
    v8h lo = *(const v8h*)p;
    v8h hi = *(const v8h*)(p + 16);
    v16h r;
#pragma unroll
    for (int i = 0; i < 8; ++i) { r[i] = lo[i]; r[8 + i] = hi[i]; }
    return r;
}

__device__ __forceinline__ float sigmoidf_(float x) {
    return 1.0f / (1.0f + __expf(-x));
}

// ---------------------------------------------------------------------------
// Setup kernels
// ---------------------------------------------------------------------------
__global__ void k_adj(const int* __restrict__ esrc, const int* __restrict__ edst,
                      int* __restrict__ adj, int* __restrict__ cnt) {
    int n = threadIdx.x;            // 64 threads, one per node
    if (n >= N) return;
    int c = 0;
    for (int e = 0; e < E; ++e)
        if (edst[e] == n) adj[n * E + c++] = esrc[e];
    cnt[n] = c;
}

// f32 -> f16 bulk convert (grid-stride)
__global__ void k_cvt(const float* __restrict__ src, _Float16* __restrict__ dst,
                      int n) {
    int stride = gridDim.x * 256;
    for (int i = blockIdx.x * 256 + threadIdx.x; i < n; i += stride)
        dst[i] = (_Float16)src[i];
}

// Wg (EMB,H) f32 -> WgT (H,EMB) f16 so GCN B-frags are K-contiguous.
__global__ void k_trh(const float* __restrict__ Wg, _Float16* __restrict__ WgT) {
    int idx = blockIdx.x * 256 + threadIdx.x;   // EMB*H elements
    int f = idx / H, h = idx % H;
    WgT[(size_t)h * EMB + f] = (_Float16)Wg[idx];
}

// ---------------------------------------------------------------------------
// Kernel 1: e0 = x[:,:,0,:] @ W_init[n]^T + b_init      (B,N,EMB) f32
// grid (N, B/16), 256 threads (8 waves): wave w -> EMB tile w.
// ---------------------------------------------------------------------------
__global__ void k_init(const float* __restrict__ x,
                       const _Float16* __restrict__ Wi16,
                       const float* __restrict__ b_init, float* __restrict__ e) {
    int n = blockIdx.x, bt = blockIdx.y;
    int wave = threadIdx.x >> 5, lane = threadIdx.x & 31;
    const float* xa = x + ((size_t)(bt * 16) * N + n) * (size_t)(T * D);  // t=0
    const size_t xstride = (size_t)N * T * D;
    v16h a0 = load_a_f32(xa, xstride, 0);
    v16h a1 = load_a_f32(xa, xstride, 32);
    const _Float16* wb = Wi16 + (size_t)n * EMB * D;
    int f0 = wave * 16;
    v8f c = {};
    c = wmma_f16(a0, load_b_f16(wb, D, f0, 0),  c);
    c = wmma_f16(a1, load_b_f16(wb, D, f0, 32), c);
    int colL = lane & 15;
    float bias = b_init[n * EMB + f0 + colL];
#pragma unroll
    for (int v = 0; v < 8; ++v) {
        int m = v + ((lane >> 4) << 3);
        e[((size_t)(bt * 16 + m) * N + n) * EMB + f0 + colL] = c[v] + bias;
    }
}

// ---------------------------------------------------------------------------
// Kernel 2 (per step): GCN  ge = tanh(((e + sum_in e[src]) / deg) @ Wg + bg)
// grid (N, B/16), 256 threads; output ge16 (B,N,H) f16.
// ---------------------------------------------------------------------------
__global__ void k_gcn(const float* __restrict__ e,
                      const _Float16* __restrict__ WgT16,
                      const float* __restrict__ bg, const int* __restrict__ adj,
                      const int* __restrict__ cnt, _Float16* __restrict__ ge16) {
    constexpr int LDA = EMB + 16;                 // 16B-aligned rows
    __shared__ _Float16 aggS[16 * LDA];
    int n = blockIdx.x, bt = blockIdx.y;
    int tid = threadIdx.x;
    int c = cnt[n];
    float inv = 1.0f / (float)(c + 1);

    {   // aggregate: 16 rows x 128 cols; each thread: fixed col, 8 rows
        int col = tid & 127;
        int r0  = (tid >> 7) * 8;
        float acc[8];
#pragma unroll
        for (int r = 0; r < 8; ++r)
            acc[r] = e[((size_t)(bt * 16 + r0 + r) * N + n) * EMB + col];
        for (int j = 0; j < c; ++j) {
            int src = adj[n * E + j];
#pragma unroll
            for (int r = 0; r < 8; ++r)
                acc[r] += e[((size_t)(bt * 16 + r0 + r) * N + src) * EMB + col];
        }
#pragma unroll
        for (int r = 0; r < 8; ++r)
            aggS[(r0 + r) * LDA + col] = (_Float16)(acc[r] * inv);
    }
    __syncthreads();

    int wave = tid >> 5, lane = tid & 31;
    v16h a[4];
#pragma unroll
    for (int kt = 0; kt < 4; ++kt) a[kt] = load_a_lds(aggS, LDA, kt * 32);

    int colL = lane & 15;
#pragma unroll
    for (int i = 0; i < 2; ++i) {
        int h0 = (wave * 2 + i) * 16;            // 16 H-tiles over 8 waves
        v8f acc = {};
#pragma unroll
        for (int kt = 0; kt < 4; ++kt)
            acc = wmma_f16(a[kt], load_b_f16(WgT16, EMB, h0, kt * 32), acc);
        float bias = bg[h0 + colL];
#pragma unroll
        for (int v = 0; v < 8; ++v) {
            int m = v + ((lane >> 4) << 3);
            ge16[((size_t)(bt * 16 + m) * N + n) * H + h0 + colL] =
                (_Float16)tanhf(acc[v] + bias);
        }
    }
}

// ---------------------------------------------------------------------------
// Kernel 3 (per step): per-node GRU cell + last-layer, writes e (and out when
// t == final_index[b]-1).  grid (N, B/16), 256 threads (8 waves).
// ---------------------------------------------------------------------------
__global__ void k_gru(const float* __restrict__ x,
                      const _Float16* __restrict__ ge16,
                      const _Float16* __restrict__ Wih16,
                      const _Float16* __restrict__ Whh16,
                      const float* __restrict__ b_ih, const float* __restrict__ b_hh,
                      const _Float16* __restrict__ Wl16,
                      const float* __restrict__ b_last,
                      const int* __restrict__ fin, float* __restrict__ e,
                      float* __restrict__ out, int t) {
    constexpr int LDH = H + 16;                   // 16B-aligned rows
    __shared__ _Float16 hS[16 * LDH];
    int n = blockIdx.x, bt = blockIdx.y;
    int wave = threadIdx.x >> 5, lane = threadIdx.x & 31;
    int colL = lane & 15;

    // A fragments: x_t (16x64, f32 src) and ge (16x256, f16 src)
    const float* xa = x + ((size_t)(bt * 16) * N + n) * (size_t)(T * D)
                        + (size_t)t * D;
    const size_t xstride = (size_t)N * T * D;
    v16h ax[2];
    ax[0] = load_a_f32(xa, xstride, 0);
    ax[1] = load_a_f32(xa, xstride, 32);

    const _Float16* gea = ge16 + ((size_t)(bt * 16) * N + n) * H;
    const size_t gestride = (size_t)N * H;
    v16h ah[8];
#pragma unroll
    for (int kt = 0; kt < 8; ++kt) ah[kt] = load_a_f16(gea, gestride, kt * 32);

    const _Float16* wih = Wih16 + (size_t)n * G3H * D;
    const _Float16* whh = Whh16 + (size_t)n * G3H * H;

    for (int i = 0; i < 2; ++i) {
        int jt = wave + i * 8;                   // h-tile 0..15
        int h0 = jt * 16;
        v8f aR = {}, aZ = {}, aNx = {}, aNh = {};
#pragma unroll
        for (int kt = 0; kt < 2; ++kt) {         // x contributions (K=64)
            aR  = wmma_f16(ax[kt], load_b_f16(wih, D,           h0, kt * 32), aR);
            aZ  = wmma_f16(ax[kt], load_b_f16(wih, D,     H  +  h0, kt * 32), aZ);
            aNx = wmma_f16(ax[kt], load_b_f16(wih, D, 2 * H  +  h0, kt * 32), aNx);
        }
#pragma unroll
        for (int kt = 0; kt < 8; ++kt) {         // hidden contributions (K=256)
            aR  = wmma_f16(ah[kt], load_b_f16(whh, H,           h0, kt * 32), aR);
            aZ  = wmma_f16(ah[kt], load_b_f16(whh, H,     H  +  h0, kt * 32), aZ);
            aNh = wmma_f16(ah[kt], load_b_f16(whh, H, 2 * H  +  h0, kt * 32), aNh);
        }
        float biR  = b_ih[n * G3H +          h0 + colL] + b_hh[n * G3H +          h0 + colL];
        float biZ  = b_ih[n * G3H +     H +  h0 + colL] + b_hh[n * G3H +     H +  h0 + colL];
        float biNx = b_ih[n * G3H + 2 * H +  h0 + colL];
        float biNh = b_hh[n * G3H + 2 * H +  h0 + colL];
#pragma unroll
        for (int v = 0; v < 8; ++v) {
            int m = v + ((lane >> 4) << 3);
            float gev = (float)ge16[((size_t)(bt * 16 + m) * N + n) * H + h0 + colL];
            float r  = sigmoidf_(aR[v] + biR);
            float z  = sigmoidf_(aZ[v] + biZ);
            float nn = tanhf(aNx[v] + biNx + r * (aNh[v] + biNh));
            float hv = (1.0f - z) * nn + z * gev;
            hS[m * LDH + h0 + colL] = (_Float16)hv;
        }
    }
    __syncthreads();

    // e_new = tanh(h @ W_last[n]^T + b_last): wave w -> EMB tile w
    int f0 = wave * 16;
    const _Float16* wl = Wl16 + (size_t)n * EMB * H;
    v8f acc = {};
#pragma unroll
    for (int kt = 0; kt < 8; ++kt) {
        v16h a = load_a_lds(hS, LDH, kt * 32);
        acc = wmma_f16(a, load_b_f16(wl, H, f0, kt * 32), acc);
    }
    float bias = b_last[n * EMB + f0 + colL];
#pragma unroll
    for (int v = 0; v < 8; ++v) {
        int m = v + ((lane >> 4) << 3);
        int brow = bt * 16 + m;
        float ev = tanhf(acc[v] + bias);
        size_t idx = ((size_t)brow * N + n) * EMB + f0 + colL;
        e[idx] = ev;
        if (fin[brow] == t + 1) out[idx] = ev;
    }
}

// ---------------------------------------------------------------------------
extern "C" void kernel_launch(void* const* d_in, const int* in_sizes, int n_in,
                              void* d_out, int out_size, void* d_ws, size_t ws_size,
                              hipStream_t stream) {
    const float* x      = (const float*)d_in[0];
    const int*   fin    = (const int*)  d_in[1];
    const int*   esrc   = (const int*)  d_in[2];
    const int*   edst   = (const int*)  d_in[3];
    const float* W_init = (const float*)d_in[4];
    const float* b_init = (const float*)d_in[5];
    const float* Wg     = (const float*)d_in[6];
    const float* bg     = (const float*)d_in[7];
    const float* W_ih   = (const float*)d_in[8];
    const float* W_hh   = (const float*)d_in[9];
    const float* b_ih   = (const float*)d_in[10];
    const float* b_hh   = (const float*)d_in[11];
    const float* W_last = (const float*)d_in[12];
    const float* b_last = (const float*)d_in[13];
    float* out = (float*)d_out;

    // workspace carve-up (~39 MB); every f16 array offset is a multiple of
    // 8 halves -> 16B alignment preserved for v8h loads.
    float*    e     = (float*)d_ws;                      // B*N*EMB f32
    _Float16* ge16  = (_Float16*)(e + (size_t)B * N * EMB); // B*N*H
    _Float16* WgT16 = ge16  + (size_t)B * N * H;         // H*EMB
    _Float16* Wi16  = WgT16 + (size_t)H * EMB;           // N*EMB*D
    _Float16* Wih16 = Wi16  + (size_t)N * EMB * D;       // N*3H*D
    _Float16* Whh16 = Wih16 + (size_t)N * G3H * D;       // N*3H*H
    _Float16* Wl16  = Whh16 + (size_t)N * G3H * H;       // N*EMB*H
    int*      adj   = (int*)(Wl16 + (size_t)N * EMB * H);// N*E
    int*      cnt   = adj + (size_t)N * E;               // N

    k_adj<<<1, 64, 0, stream>>>(esrc, edst, adj, cnt);
    k_trh<<<(EMB * H) / 256, 256, 0, stream>>>(Wg, WgT16);
    k_cvt<<<512, 256, 0, stream>>>(W_init, Wi16, N * EMB * D);
    k_cvt<<<1024, 256, 0, stream>>>(W_ih, Wih16, N * G3H * D);
    k_cvt<<<2048, 256, 0, stream>>>(W_hh, Whh16, N * G3H * H);
    k_cvt<<<1024, 256, 0, stream>>>(W_last, Wl16, N * EMB * H);

    dim3 grid(N, B / 16);
    k_init<<<grid, 256, 0, stream>>>(x, Wi16, b_init, e);

    for (int t = 0; t < T; ++t) {
        k_gcn<<<grid, 256, 0, stream>>>(e, WgT16, bg, adj, cnt, ge16);
        k_gru<<<grid, 256, 0, stream>>>(x, ge16, Wih16, Whh16, b_ih, b_hh,
                                        Wl16, b_last, fin, e, out, t);
    }
}
